// VQVAE_79070347919596
// MI455X (gfx1250) — matrix operand order repaired and checked by reading, not compiled, not run on hardware
//
#include <hip/hip_runtime.h>
#include <hip/hip_bf16.h>

// ---------------------------------------------------------------------------
// VQVAE forward for MI455X (gfx1250, wave32, WMMA).
// Channels-last activations make k3-conv im2col a contiguous read; every conv
// and the vocab projection run as bf16 WMMA GEMMs (f32 accumulate) with a
// double-buffered register pipeline. VQ argmin is a VALU/LDS kernel.
// ---------------------------------------------------------------------------

typedef __bf16 bf16;
typedef __attribute__((ext_vector_type(16))) __bf16 v16bf;
typedef __attribute__((ext_vector_type(8)))  float  v8f;
typedef __attribute__((ext_vector_type(4)))  float  f32x4;

struct Pair128 { f32x4 lo; f32x4 hi; };

static __device__ __forceinline__ v16bf load_frag(const bf16* p0, const bf16* p1) {
  Pair128 t;
  t.lo = *reinterpret_cast<const f32x4*>(p0);
  t.hi = *reinterpret_cast<const f32x4*>(p1);
  return __builtin_bit_cast(v16bf, t);
}

constexpr int Bc = 16, Lc = 2048, Ec = 128, Hc = 256, Dc = 64, Vc = 4096;
constexpr int BL = Bc * Lc;        // 32768 positions
constexpr int LP = Lc + 2;         // padded length for k3 convs

// ---------------------------------------------------------------------------
// Generic WMMA GEMM: out[p, n] = relu?(sum_k A[p, k] * W[n, k] + bias[n])
// One wave: 16 positions x (NT*16) columns; K double-buffered (ksteps even).
// ---------------------------------------------------------------------------
template <int NT, bool RELU, bool OUTF32>
__global__ __launch_bounds__(256) void gemm_ws(
    const bf16* __restrict__ A, const bf16* __restrict__ W,
    const float* __restrict__ bias, void* __restrict__ Out,
    int inC, int inPad, int Kdim,
    int outC, int outPad, int outBase, int nChunks)
{
  const int lane    = threadIdx.x & 31;
  const int wid     = blockIdx.x * 8 + (threadIdx.x >> 5);
  const int rowTile = wid / nChunks;
  const int nChunk  = wid % nChunks;
  const int mrow    = lane & 15;
  const int hi      = lane >> 4;

  const int p = rowTile * 16 + mrow;
  // A fragment base: 16x32 bf16 tile; lane half `hi` covers K octets {hi*8, 16+hi*8}
  const bf16* aq = A + (size_t)p * inC + (size_t)(p / Lc) * inPad + hi * 8;
  const int ncolBase = nChunk * (NT * 16);
  // B fragment base: lane holds column (ncolBase + mrow + nt*16), K octets {hi*8, 16+hi*8}
  const bf16* wq = W + (size_t)(ncolBase + mrow) * Kdim + hi * 8;
  const size_t wstride = (size_t)16 * Kdim;

  v8f acc[NT];
#pragma unroll
  for (int nt = 0; nt < NT; ++nt)
#pragma unroll
    for (int r = 0; r < 8; ++r) acc[nt][r] = 0.0f;

  const int ksteps = Kdim >> 5;   // even for every layer in this model

  v16bf aX, aY, bX[NT], bY[NT];
  aX = load_frag(aq, aq + 16);
#pragma unroll
  for (int nt = 0; nt < NT; ++nt) {
    const bf16* w0 = wq + nt * wstride;
    bX[nt] = load_frag(w0, w0 + 16);
  }

  for (int kt = 0; kt < ksteps; kt += 2) {
    {   // stage kt+1 while computing kt
      const int kb = (kt + 1) * 32;
      aY = load_frag(aq + kb, aq + kb + 16);
#pragma unroll
      for (int nt = 0; nt < NT; ++nt) {
        const bf16* w1 = wq + nt * wstride + kb;
        bY[nt] = load_frag(w1, w1 + 16);
      }
    }
#pragma unroll
    for (int nt = 0; nt < NT; ++nt)
      acc[nt] = __builtin_amdgcn_wmma_f32_16x16x32_bf16(
          false, aX, false, bX[nt], (short)0, acc[nt], false, false);

    if (kt + 2 < ksteps) {   // stage kt+2 while computing kt+1
      const int kb = (kt + 2) * 32;
      aX = load_frag(aq + kb, aq + kb + 16);
#pragma unroll
      for (int nt = 0; nt < NT; ++nt) {
        const bf16* w2 = wq + nt * wstride + kb;
        bX[nt] = load_frag(w2, w2 + 16);
      }
    }
#pragma unroll
    for (int nt = 0; nt < NT; ++nt)
      acc[nt] = __builtin_amdgcn_wmma_f32_16x16x32_bf16(
          false, aY, false, bY[nt], (short)0, acc[nt], false, false);
  }

#pragma unroll
  for (int nt = 0; nt < NT; ++nt) {
    const int col = ncolBase + nt * 16 + mrow;
    const float bv = bias[col];
#pragma unroll
    for (int r = 0; r < 8; ++r) {
      const int pm = rowTile * 16 + hi * 8 + r;   // C/D layout: M = r + hi*8
      float v = acc[nt][r] + bv;
      if (RELU) v = fmaxf(v, 0.0f);
      const size_t o = (size_t)pm * outC + (size_t)(pm / Lc) * outPad + outBase + col;
      if (OUTF32) reinterpret_cast<float*>(Out)[o] = v;
      else        reinterpret_cast<bf16*>(Out)[o] = (bf16)v;
    }
  }
}

// ---------------------------------------------------------------------------
// Weight repack: (Co,Ci,Kt) f32 -> bf16 [co][k*Ci+ci]  (Kt=1 => plain convert)
// ---------------------------------------------------------------------------
__global__ __launch_bounds__(256) void repack_kernel(
    const float* __restrict__ w, bf16* __restrict__ out, int Co, int Ci, int Kt)
{
  const int i = blockIdx.x * 256 + threadIdx.x;
  const int n = Co * Ci * Kt;
  if (i >= n) return;
  const int co  = i / (Ci * Kt);
  const int rem = i % (Ci * Kt);
  const int ci  = rem / Kt;
  const int k   = rem % Kt;
  out[(size_t)co * (Ci * Kt) + k * Ci + ci] = (bf16)w[i];
}

// ---------------------------------------------------------------------------
// Embedding gather into padded channels-last bf16 buffer (pad rows pre-zeroed)
// ---------------------------------------------------------------------------
__global__ __launch_bounds__(256) void embed_kernel(
    const int* __restrict__ x, const float* __restrict__ te, bf16* __restrict__ out)
{
  const int i = blockIdx.x * 256 + threadIdx.x;   // i < BL*Ec
  const int p = i >> 7;                           // /Ec
  const int e = i & (Ec - 1);
  const int tok = x[p];
  const float v = te[(size_t)tok * Ec + e];
  out[(size_t)p * Ec + (size_t)(p >> 11) * (2 * Ec) + Ec + e] = (bf16)v;
}

// ---------------------------------------------------------------------------
// VQ: per-position argmin over 512 codes of |c|^2 - 2 z.c, codebook staged
// through LDS in 16 KB chunks; emits codes, z_q (bf16), commit-loss partials.
// ---------------------------------------------------------------------------
__global__ __launch_bounds__(256) void vq_kernel(
    const float* __restrict__ z_e, const float* __restrict__ cb,
    bf16* __restrict__ z_q, int* __restrict__ codes, float* __restrict__ lossAcc)
{
  __shared__ float scb[64 * 64];
  __shared__ float red[256];
  const int tid = threadIdx.x;
  const int p = blockIdx.x * 256 + tid;

  float z[64];
#pragma unroll
  for (int d = 0; d < 64; ++d) z[d] = z_e[(size_t)p * 64 + d];

  float best = 3.4e38f;
  int bi = 0;
  for (int ch = 0; ch < 8; ++ch) {
    __syncthreads();
#pragma unroll
    for (int i = 0; i < 16; ++i)
      scb[tid * 16 + i] = cb[ch * 4096 + tid * 16 + i];
    __syncthreads();
    for (int c = 0; c < 64; ++c) {
      float s = 0.0f;
#pragma unroll
      for (int d = 0; d < 64; ++d) {
        const float t = scb[c * 64 + d];
        s = fmaf(t, t - 2.0f * z[d], s);      // |c|^2 - 2 z.c
      }
      if (s < best) { best = s; bi = ch * 64 + c; }
    }
  }
  codes[p] = bi;

  float sq = 0.0f;
#pragma unroll
  for (int d = 0; d < 64; ++d) {
    const float c = cb[(size_t)bi * 64 + d];
    const float df = z[d] - c;
    sq = fmaf(df, df, sq);
    z_q[(size_t)p * 64 + d] = (bf16)c;
  }

  __syncthreads();
  red[tid] = sq;
  __syncthreads();
  for (int s2 = 128; s2 > 0; s2 >>= 1) {
    if (tid < s2) red[tid] += red[tid + s2];
    __syncthreads();
  }
  if (tid == 0) atomicAdd(lossAcc, red[0]);
}

// ---------------------------------------------------------------------------
// Finalize: loss scalar + codes (as float) appended after logits in d_out.
// ---------------------------------------------------------------------------
__global__ __launch_bounds__(256) void finalize_kernel(
    const float* __restrict__ lossAcc, const int* __restrict__ codes,
    float* __restrict__ out)
{
  const size_t base = (size_t)BL * Vc;
  const int i = blockIdx.x * 256 + threadIdx.x;
  if (i == 0) out[base] = 0.1f * lossAcc[0] / (float)(BL * Dc);
  if (i < BL) out[base + 1 + i] = (float)codes[i];
}

// ---------------------------------------------------------------------------
extern "C" void kernel_launch(void* const* d_in, const int* in_sizes, int n_in,
                              void* d_out, int out_size, void* d_ws, size_t ws_size,
                              hipStream_t stream)
{
  (void)in_sizes; (void)n_in; (void)out_size; (void)ws_size;
  const int*   x         = (const int*)  d_in[0];
  const float* token_emb = (const float*)d_in[1];
  const float* ew1 = (const float*)d_in[2];  const float* eb1 = (const float*)d_in[3];
  const float* ew2 = (const float*)d_in[4];  const float* eb2 = (const float*)d_in[5];
  const float* ew3 = (const float*)d_in[6];  const float* eb3 = (const float*)d_in[7];
  const float* codebook = (const float*)d_in[8];
  const float* dw1 = (const float*)d_in[9];  const float* db1 = (const float*)d_in[10];
  const float* dw2 = (const float*)d_in[11]; const float* db2 = (const float*)d_in[12];
  const float* dw3 = (const float*)d_in[13]; const float* db3 = (const float*)d_in[14];
  const float* out_w = (const float*)d_in[15]; const float* out_b = (const float*)d_in[16];
  float* out = (float*)d_out;

  // ---- workspace carve (256 B aligned) ----
  char* wsp = (char*)d_ws; size_t off = 0;
  auto alloc = [&](size_t bytes) -> void* {
    void* p = wsp + off; off = (off + bytes + 255) & ~(size_t)255; return p;
  };
  bf16* Wf1  = (bf16*)alloc((size_t)Hc * Ec * 3 * 2);
  bf16* Wf2  = (bf16*)alloc((size_t)Hc * Hc * 3 * 2);
  bf16* Wf3  = (bf16*)alloc((size_t)Dc * Hc * 2);
  bf16* Wd1  = (bf16*)alloc((size_t)Hc * Dc * 2);
  bf16* Wd2  = (bf16*)alloc((size_t)Hc * Hc * 3 * 2);
  bf16* Wd3  = (bf16*)alloc((size_t)Ec * Hc * 2);
  bf16* Wout = (bf16*)alloc((size_t)Vc * Ec * 2);
  bf16* emb_pad = (bf16*)alloc((size_t)Bc * LP * Ec * 2);
  bf16* h1_pad  = (bf16*)alloc((size_t)Bc * LP * Hc * 2);
  bf16* h2      = (bf16*)alloc((size_t)BL * Hc * 2);
  float* z_e    = (float*)alloc((size_t)BL * Dc * 4);
  bf16* z_q     = (bf16*)alloc((size_t)BL * Dc * 2);
  bf16* d1_pad  = (bf16*)alloc((size_t)Bc * LP * Hc * 2);
  bf16* d2      = (bf16*)alloc((size_t)BL * Hc * 2);
  bf16* d3      = (bf16*)alloc((size_t)BL * Ec * 2);
  int*  codes   = (int*) alloc((size_t)BL * 4);
  float* lossAcc = (float*)alloc(256);

  // ---- zero pad rows (full buffers; cheap) + loss accumulator ----
  hipMemsetAsync(emb_pad, 0, (size_t)Bc * LP * Ec * 2, stream);
  hipMemsetAsync(h1_pad,  0, (size_t)Bc * LP * Hc * 2, stream);
  hipMemsetAsync(d1_pad,  0, (size_t)Bc * LP * Hc * 2, stream);
  hipMemsetAsync(lossAcc, 0, 4, stream);

  // ---- weight repack (fp32 -> bf16, [co][k*Ci+ci]) ----
  auto rp = [&](const float* w, bf16* o, int Co, int Ci, int Kt) {
    const int n = Co * Ci * Kt;
    repack_kernel<<<(n + 255) / 256, 256, 0, stream>>>(w, o, Co, Ci, Kt);
  };
  rp(ew1,   Wf1,  Hc, Ec, 3);
  rp(ew2,   Wf2,  Hc, Hc, 3);
  rp(ew3,   Wf3,  Dc, Hc, 1);
  rp(dw1,   Wd1,  Hc, Dc, 1);
  rp(dw2,   Wd2,  Hc, Hc, 3);
  rp(dw3,   Wd3,  Ec, Hc, 1);
  rp(out_w, Wout, Vc, Ec, 1);

  // ---- embedding gather ----
  embed_kernel<<<(BL * Ec) / 256, 256, 0, stream>>>(x, token_emb, emb_pad);

  constexpr int rowTiles = BL / 16;   // 2048

  // ---- encoder ----
  // conv1 (k3, E->H, relu): Kdim=384, out=h1_pad (padded); N=256 -> 4 chunks
  gemm_ws<4, true, false><<<rowTiles * 4 / 8, 256, 0, stream>>>(
      emb_pad, Wf1, eb1, h1_pad, Ec, 2 * Ec, 3 * Ec, Hc, 2 * Hc, Hc, 4);
  // conv2 (k3, H->H, relu): Kdim=768, out=h2
  gemm_ws<4, true, false><<<rowTiles * 4 / 8, 256, 0, stream>>>(
      h1_pad, Wf2, eb2, h2, Hc, 2 * Hc, 3 * Hc, Hc, 0, 0, 4);
  // conv3 (k1, H->D, no relu, f32 out = z_e): N=64 -> 1 chunk
  gemm_ws<4, false, true><<<rowTiles / 8, 256, 0, stream>>>(
      h2, Wf3, eb3, z_e, Hc, 0, Hc, Dc, 0, 0, 1);

  // ---- vector quantization ----
  vq_kernel<<<BL / 256, 256, 0, stream>>>(z_e, codebook, z_q, codes, lossAcc);

  // ---- decoder ----
  // dconv1 (k1, D->H, relu): Kdim=64, out=d1_pad (padded)
  gemm_ws<4, true, false><<<rowTiles * 4 / 8, 256, 0, stream>>>(
      z_q, Wd1, db1, d1_pad, Dc, 0, Dc, Hc, 2 * Hc, Hc, 4);
  // dconv2 (k3, H->H, relu): Kdim=768, out=d2
  gemm_ws<4, true, false><<<rowTiles * 4 / 8, 256, 0, stream>>>(
      d1_pad, Wd2, db2, d2, Hc, 2 * Hc, 3 * Hc, Hc, 0, 0, 4);
  // dconv3 (k1, H->E, relu): N=128 -> 2 chunks
  gemm_ws<4, true, false><<<rowTiles * 2 / 8, 256, 0, stream>>>(
      d2, Wd3, db3, d3, Hc, 0, Hc, Ec, 0, 0, 2);

  // ---- output projection (E->V, f32 logits straight to d_out): 64 chunks ----
  gemm_ws<4, false, true><<<rowTiles * 64 / 8, 256, 0, stream>>>(
      d3, Wout, out_b, out, Ec, 0, Ec, Vc, 0, 0, 64);

  // ---- loss + codes ----
  finalize_kernel<<<BL / 256, 256, 0, stream>>>(lossAcc, codes, out);
}
